// GnCritic_5849745457780
// MI455X (gfx1250) — matrix operand order repaired and checked by reading, not compiled, not run on hardware
//
#include <hip/hip_runtime.h>
#include <hip/hip_bf16.h>

typedef _Float16 half_t;
typedef __attribute__((ext_vector_type(16))) _Float16 v16h;
typedef __attribute__((ext_vector_type(8)))  _Float16 v8h;
typedef __attribute__((ext_vector_type(8)))  float    v8f;

// ---------------- weight scratch layout (f16, transposed [N][Kpad]) ----------------
#define WS_MP   0        // mp_w : N=128, Kpad=64  (K=46)   -> 8192 halves
#define WS_W1   8192     // phi_w1: N=256, Kpad=160 (K=157) -> 40960
#define WS_W3   49152    // phi_w3: same                         40960
#define WS_W2   90112    // phi_w2: N=256, Kpad=256              65536
#define WS_W4   155648   // phi_w4: same                         65536
#define WS_TOT  221184   // halves (442368 bytes)

// ---------------- LDS layout (bytes) — lifetime-overlapped ----------------
// Region 0: edge-A (phases 0-2) then hidden h (phases 4+)
#define L_EA    0        // 320 x 64  f16 = 40960
#define L_H     0        // 80  x 256 f16 = 40960   (alias, disjoint lifetime)
// Region 1: agg (phases 1-3) then o + reduction scratch (phases 5+)
#define L_AGG   40960    // 16x5x128 f32 = 40960
#define L_O     40960    // 16 x 256 f32 = 16384    (alias)
#define L_RED   57344    // 16 x 16  f32 = 1024     (alias)
// Persistent:
#define L_PA    81920    // phi A : 80 x 160 f16 = 25600
#define L_OBS   107520   // 16 x 85 f32 = 5440
#define L_ACT   112960   // 16 x 4  f32 = 256
#define L_DG    113216   // 16 x 30 f32 = 1920
#define L_TOT   115200   // ~112.5 KB -> 2 workgroups per WGP (320 KB LDS)

// ------------------------------------------------------------------
// Prep: convert all weights to f16, transposed to [N][Kpad], K zero-padded.
// ------------------------------------------------------------------
__global__ void prep_weights(const float* __restrict__ mp_w,
                             const float* __restrict__ w1,
                             const float* __restrict__ w3,
                             const float* __restrict__ w2,
                             const float* __restrict__ w4,
                             half_t* __restrict__ ws) {
  int idx = blockIdx.x * blockDim.x + threadIdx.x;
  if (idx >= WS_TOT) return;
  if (idx < WS_W1) {                       // mp: K=46->64, N=128
    int j = idx - WS_MP; int n = j >> 6, k = j & 63;
    ws[idx] = (k < 46) ? (half_t)mp_w[k * 128 + n] : (half_t)0.f;
  } else if (idx < WS_W3) {                // w1: K=157->160, N=256
    int j = idx - WS_W1; int n = j / 160, k = j % 160;
    ws[idx] = (k < 157) ? (half_t)w1[k * 256 + n] : (half_t)0.f;
  } else if (idx < WS_W2) {                // w3
    int j = idx - WS_W3; int n = j / 160, k = j % 160;
    ws[idx] = (k < 157) ? (half_t)w3[k * 256 + n] : (half_t)0.f;
  } else if (idx < WS_W4) {                // w2: 256x256
    int j = idx - WS_W2; int n = j >> 8, k = j & 255;
    ws[idx] = (half_t)w2[k * 256 + n];
  } else {                                 // w4
    int j = idx - WS_W4; int n = j >> 8, k = j & 255;
    ws[idx] = (half_t)w4[k * 256 + n];
  }
}

// ------------------------------------------------------------------
// Fragment loaders (WMMA 16x16x32 f16, wave32; contiguous k = 16*(lane/16) + s, same in A and B)
// ------------------------------------------------------------------
// A (16x32, MxK): lanes 0-15 row M=lane, halves = K[kb..kb+15],
// kb = 0 for lanes<16, 16 for lanes>=16.
__device__ __forceinline__ v16h load_a_lds(const half_t* base, int strideK,
                                           int r0, int k0, int lane) {
  int row = r0 + (lane & 15);
  int kb  = (lane < 16) ? 0 : 16;
  const half_t* p = base + row * strideK + k0 + kb;
  v8h lo = *(const v8h*)(p);
  v8h hi = *(const v8h*)(p + 8);
  v16h a;
#pragma unroll
  for (int i = 0; i < 8; ++i) { a[i] = lo[i]; a[i + 8] = hi[i]; }
  return a;
}

// B (32x16, KxN) from transposed weights W[n][Kpad]:
// lane<16: col N=lane, K = k0..k0+15 ; lane>=16: col N=lane-16, K = k0+16..k0+31.
__device__ __forceinline__ v16h load_b_glb(const half_t* __restrict__ W, int Kpad,
                                           int n0, int k0, int lane) {
  int n  = n0 + (lane & 15);
  int kb = (lane < 16) ? 0 : 16;
  return *(const v16h*)(W + n * Kpad + k0 + kb);
}

__device__ __forceinline__ v8f wmma16(v16h a, v16h b, v8f c) {
  return __builtin_amdgcn_wmma_f32_16x16x32_f16(false, a, false, b,
                                                (short)0, c, false, false);
}

// ------------------------------------------------------------------
// Fused critic: one block = 16 batch rows, 256 threads (8 waves)
// ------------------------------------------------------------------
__global__ __launch_bounds__(256)
void gn_critic_fused(const float* __restrict__ obs,
                     const float* __restrict__ act,
                     const float* __restrict__ ag,
                     const float* __restrict__ g,
                     const float* __restrict__ mp_b,
                     const float* __restrict__ phi_b1,
                     const float* __restrict__ phi_b2,
                     const float* __restrict__ phi_b3,
                     const float* __restrict__ phi_b4,
                     const float* __restrict__ rho_w1,
                     const float* __restrict__ rho_b1,
                     const float* __restrict__ rho_w2,
                     const float* __restrict__ rho_b2,
                     const int* __restrict__ edges_src,
                     const int* __restrict__ edges_dst,
                     const int* __restrict__ pred_ids,
                     const half_t* __restrict__ ws,
                     float* __restrict__ out, int B) {
  extern __shared__ __align__(16) char smem[];
  half_t* sEA  = (half_t*)(smem + L_EA);
  half_t* sH   = (half_t*)(smem + L_H);
  float*  sAgg = (float*)(smem + L_AGG);
  float*  sO   = (float*)(smem + L_O);
  float*  sRed = (float*)(smem + L_RED);
  half_t* sPA  = (half_t*)(smem + L_PA);
  float*  sOBS = (float*)(smem + L_OBS);
  float*  sACT = (float*)(smem + L_ACT);
  float*  sDG  = (float*)(smem + L_DG);

  const int tid  = threadIdx.x;
  const int lane = tid & 31;
  const int wave = tid >> 5;
  const int b0   = blockIdx.x * 16;
  const int mb   = (lane < 16) ? 0 : 8;   // C/D row base for this lane half

  // ---- Phase 0: stage raw per-sample inputs ----
  for (int i = tid; i < 16 * 85; i += 256) sOBS[i] = obs[b0 * 85 + i];
  for (int i = tid; i < 16 * 4;  i += 256) sACT[i] = act[b0 * 4 + i];
  for (int i = tid; i < 16 * 30; i += 256) sDG[i]  = g[b0 * 30 + i] - ag[b0 * 30 + i];
  __syncthreads();

  // ---- Phase 1: build edge A-matrix (row = e*16+s, Kpad=64) + zero agg ----
  for (int i = tid; i < 320 * 64; i += 256) {
    int row = i >> 6, col = i & 63;
    int e = row >> 4, s = row & 15;
    float v = 0.f;
    if (col < 10)       v = sOBS[s * 85 + col];                                   // obs_body
    else if (col < 14)  v = sACT[s * 4 + (col - 10)];                             // act
    else if (col < 16)  v = sDG[s * 30 + pred_ids[e * 2 + (col - 14)]];           // dg_sel
    else if (col < 31)  v = sOBS[s * 85 + 10 + edges_src[e] * 15 + (col - 16)];   // src obj
    else if (col < 46)  v = sOBS[s * 85 + 10 + edges_dst[e] * 15 + (col - 31)];   // dst obj
    sEA[i] = (half_t)v;
  }
  for (int i = tid; i < 16 * 5 * 128; i += 256) sAgg[i] = 0.f;
  __syncthreads();

  // ---- Phase 2: edge GEMM (320x64 @ 64x128) + ReLU + aggregate by dst ----
  {
    const int n0 = wave * 16;                    // 8 waves cover N=128 exactly
    const int n  = n0 + (lane & 15);
    const float bias = mp_b[n];
    const half_t* Wmp = ws + WS_MP;
    v16h b0f = load_b_glb(Wmp, 64, n0, 0, lane); // loop-invariant B fragments
    v16h b1f = load_b_glb(Wmp, 64, n0, 32, lane);
    for (int e = 0; e < 20; ++e) {
      v8f c = {};
      c = wmma16(load_a_lds(sEA, 64, e * 16, 0, lane),  b0f, c);
      c = wmma16(load_a_lds(sEA, 64, e * 16, 32, lane), b1f, c);
      int dst = edges_dst[e];
#pragma unroll
      for (int r = 0; r < 8; ++r) {
        float v = c[r] + bias; v = v > 0.f ? v : 0.f;
        int s = mb + r;                          // sample index
        sAgg[(s * 5 + dst) * 128 + n] += v;      // wave owns this n-strip: race-free
      }
    }
  }
  __syncthreads();

  // ---- Phase 3: build phi A-matrix (row = obj*16+s, Kpad=160) ----
  for (int i = tid; i < 80 * 160; i += 256) {
    int row = i / 160, col = i % 160;
    int obj = row >> 4, s = row & 15;
    float v = 0.f;
    if (col < 4)        v = sACT[s * 4 + col];                              // act
    else if (col < 14)  v = sOBS[s * 85 + (col - 4)];                       // obs_body
    else if (col < 29)  v = sOBS[s * 85 + 10 + obj * 15 + (col - 14)];      // obs_obj
    else if (col < 157) v = sAgg[(s * 5 + obj) * 128 + (col - 29)];         // agg
    sPA[i] = (half_t)v;
  }
  __syncthreads();

  // ---- Phases 4-6 twice: the two phi/rho stacks ----
  for (int path = 0; path < 2; ++path) {
    const half_t* Wa = ws + (path ? WS_W3 : WS_W1);
    const half_t* Wb = ws + (path ? WS_W4 : WS_W2);
    const float*  ba = path ? phi_b3 : phi_b1;
    const float*  bb = path ? phi_b4 : phi_b2;
    const float*  rw = path ? rho_w2 : rho_w1;
    const float*  rb = path ? rho_b2 : rho_b1;
    float* outp = out + path * B;

    // h = relu(phiA @ Wa + ba)   (80x160 @ 160x256 -> f16 in LDS)
    // NOTE: phase-3/6 barriers guarantee sEA/sH alias switch is safe.
    for (int strip = wave; strip < 16; strip += 8) {
      int n0 = strip * 16, n = n0 + (lane & 15);
      float bias = ba[n];
      v16h bfr[5];                                // B fragments: invariant over ro
#pragma unroll
      for (int kk = 0; kk < 5; ++kk) bfr[kk] = load_b_glb(Wa, 160, n0, kk * 32, lane);
      for (int ro = 0; ro < 5; ++ro) {
        v8f c = {};
#pragma unroll
        for (int kk = 0; kk < 5; ++kk)
          c = wmma16(load_a_lds(sPA, 160, ro * 16, kk * 32, lane), bfr[kk], c);
#pragma unroll
        for (int r = 0; r < 8; ++r) {
          float v = c[r] + bias; v = v > 0.f ? v : 0.f;
          sH[(ro * 16 + mb + r) * 256 + n] = (half_t)v;
        }
      }
    }
    __syncthreads();

    // o[s] = sum_obj relu(h @ Wb + bb)   (register accumulation over 5 obj tiles)
    for (int strip = wave; strip < 16; strip += 8) {
      int n0 = strip * 16, n = n0 + (lane & 15);
      float bias = bb[n];
      v16h bfr[8];                                // B fragments: invariant over ro
#pragma unroll
      for (int kk = 0; kk < 8; ++kk) bfr[kk] = load_b_glb(Wb, 256, n0, kk * 32, lane);
      v8f acc = {};
      for (int ro = 0; ro < 5; ++ro) {
        v8f c = {};
#pragma unroll
        for (int kk = 0; kk < 8; ++kk)
          c = wmma16(load_a_lds(sH, 256, ro * 16, kk * 32, lane), bfr[kk], c);
#pragma unroll
        for (int r = 0; r < 8; ++r) {
          float v = c[r] + bias; acc[r] += (v > 0.f ? v : 0.f);
        }
      }
#pragma unroll
      for (int r = 0; r < 8; ++r) sO[(mb + r) * 256 + n] = acc[r];
    }
    __syncthreads();

    // q[s] = o[s] . rho_w + rho_b  — parallel partial sums + LDS tree combine
    {
      int s = tid >> 4, j = tid & 15;
      float p = 0.f;
#pragma unroll
      for (int nn = 0; nn < 16; ++nn) {
        int n = j * 16 + nn;
        p += sO[s * 256 + n] * rw[n];
      }
      sRed[s * 16 + j] = p;
    }
    __syncthreads();
    if (tid < 16) {
      float sum = rb[0];
#pragma unroll
      for (int j = 0; j < 16; ++j) sum += sRed[tid * 16 + j];
      outp[b0 + tid] = sum;
    }
    __syncthreads();
  }
}

// ------------------------------------------------------------------
extern "C" void kernel_launch(void* const* d_in, const int* in_sizes, int n_in,
                              void* d_out, int out_size, void* d_ws, size_t ws_size,
                              hipStream_t stream) {
  const float* obs    = (const float*)d_in[0];
  const float* act    = (const float*)d_in[1];
  const float* ag     = (const float*)d_in[2];
  const float* g      = (const float*)d_in[3];
  const float* mp_w   = (const float*)d_in[4];
  const float* mp_b   = (const float*)d_in[5];
  const float* phi_w1 = (const float*)d_in[6];
  const float* phi_b1 = (const float*)d_in[7];
  const float* phi_w2 = (const float*)d_in[8];
  const float* phi_b2 = (const float*)d_in[9];
  const float* phi_w3 = (const float*)d_in[10];
  const float* phi_b3 = (const float*)d_in[11];
  const float* phi_w4 = (const float*)d_in[12];
  const float* phi_b4 = (const float*)d_in[13];
  const float* rho_w1 = (const float*)d_in[14];
  const float* rho_b1 = (const float*)d_in[15];
  const float* rho_w2 = (const float*)d_in[16];
  const float* rho_b2 = (const float*)d_in[17];
  const int* edges_src = (const int*)d_in[18];
  const int* edges_dst = (const int*)d_in[19];
  const int* pred_ids  = (const int*)d_in[20];
  // d_in[21] (incoming) unused: aggregation by dst is equivalent.

  const int B = in_sizes[1] / 4;          // act is (B,4)
  half_t* wsH = (half_t*)d_ws;

  prep_weights<<<(WS_TOT + 255) / 256, 256, 0, stream>>>(mp_w, phi_w1, phi_w3,
                                                         phi_w2, phi_w4, wsH);
  gn_critic_fused<<<B / 16, 256, L_TOT, stream>>>(
      obs, act, ag, g, mp_b, phi_b1, phi_b2, phi_b3, phi_b4,
      rho_w1, rho_b1, rho_w2, rho_b2, edges_src, edges_dst, pred_ids,
      wsH, (float*)d_out, B);
}
